// RelationalLayersModule_85727547228490
// MI455X (gfx1250) — compile-verified, matrix operand order checked by reading, not run on hardware
//
#include <hip/hip_runtime.h>
#include <hip/hip_bf16.h>

typedef __attribute__((ext_vector_type(16))) _Float16 v16h;
typedef __attribute__((ext_vector_type(8)))  _Float16 v8h;
typedef __attribute__((ext_vector_type(8)))  float    v8f;

#define D  64
#define TD 128

// ---- A-fragment: row-major [row][k] tile, two contiguous 16B runs per lane ----
// 16-bit A layout (ISA 7.12.2): lane -> row (lane&15), k = kBase + (e>>3)*16 +
// ((lane>>4)*8) + (e&7)  => elements 0-7 and 8-15 are contiguous in k.
__device__ __forceinline__ v16h load_a_frag(const _Float16* src, int ld, int rowBase,
                                            int lane, int kBase) {
  const int m  = lane & 15;
  const int kh = (lane >> 4) << 3;
  const _Float16* p = src + (rowBase + m) * ld + kBase + kh;
  v8h lo = *(const v8h*)p;         // k = kBase+kh .. +7
  v8h hi = *(const v8h*)(p + 16);  // k = kBase+16+kh .. +7
  return __builtin_shufflevector(lo, hi, 0, 1, 2, 3, 4, 5, 6, 7, 8, 9, 10, 11, 12,
                                 13, 14, 15);
}

// ---- B-fragment: pre-swizzled fragment-order buffer; one 32B read per lane ----
__device__ __forceinline__ v16h load_b_frag(const _Float16* Wf, int frag, int lane) {
  return *(const v16h*)(Wf + ((frag << 5) + lane) * 16);
}

__device__ __forceinline__ v8f wmma_step(v16h a, v16h b, v8f c) {
  return __builtin_amdgcn_wmma_f32_16x16x32_f16(false, a, false, b, (short)0, c,
                                                false, false);
}

// ---- Prep: fp32 weights [K=128, Ncols] row-major -> f16 fragment order ----
// fragment f = nt*4 + kt ; per lane 16 contiguous halves.
__global__ __launch_bounds__(256, 1) void swizzle_weights_kernel(
    const float* __restrict__ W, _Float16* __restrict__ out, int Ncols, int nFrag,
    long strideW, long strideOut) {
  const int mat = blockIdx.y;
  const float* Wm = W + (size_t)mat * strideW;
  _Float16*   om  = out + (size_t)mat * strideOut;
  const int i = blockIdx.x * 256 + threadIdx.x;
  if (i >= nFrag * 512) return;
  const int f    = i >> 9;
  const int lane = (i >> 4) & 31;
  const int e    = i & 15;
  const int nt   = f >> 2;
  const int kt   = f & 3;
  const int n    = nt * 16 + (lane & 15);
  const int k    = kt * 32 + ((e >> 3) << 4) + ((lane >> 4) << 3) + (e & 7);
  om[i] = (_Float16)Wm[k * Ncols + n];
}

// ================= Message MLP + scatter-sum kernel =================
// grid.x = tiles of 64 pairs, grid.y = relation r ; block = 128 (4 waves)
__global__ __launch_bounds__(128, 1) void msg_scatter_kernel(
    const float* __restrict__ emb, const int* __restrict__ relIdx,
    const _Float16* __restrict__ wm1f, const float* __restrict__ bm1,
    const _Float16* __restrict__ wm2f, const float* __restrict__ bm2,
    float* __restrict__ agg, int E) {
  __shared__ __align__(16) _Float16 sW1[TD * TD];   // 32 KB fragment-order
  __shared__ __align__(16) _Float16 sW2[TD * TD];   // 32 KB fragment-order
  __shared__ float sB1[TD];
  __shared__ float sB2[TD];
  __shared__ __align__(16) _Float16 sPair[64 * TD]; // 16 KB row-major
  __shared__ __align__(16) _Float16 sH[64 * TD];    // 16 KB row-major

  const int r    = blockIdx.y;
  const int tid  = threadIdx.x;
  const int lane = tid & 31;
  const int wave = tid >> 5;

  // coalesced 16B copy of pre-swizzled f16 weights into LDS
  {
    const uint4* g1 = (const uint4*)(wm1f + (size_t)r * TD * TD);
    const uint4* g2 = (const uint4*)(wm2f + (size_t)r * TD * TD);
    uint4* s1 = (uint4*)sW1;
    uint4* s2 = (uint4*)sW2;
#pragma unroll
    for (int i = tid; i < TD * TD / 8; i += 128) {
      s1[i] = g1[i];
      s2[i] = g2[i];
    }
  }
  if (tid < TD) {
    sB1[tid] = bm1[r * TD + tid];
    sB2[tid] = bm2[r * TD + tid];
  }

  const int  pairBase = blockIdx.x * 64;
  const int* idx      = relIdx + (size_t)r * 2 * E;

  // cooperative gather: pair p = concat(emb[idx[2p]], emb[idx[2p+1]]) -> f16 LDS
  for (int i = tid; i < 64 * 32; i += 128) {
    const int p  = i >> 5;
    const int c0 = (i & 31) << 2;
    const int pe = pairBase + p;
    float4 v = make_float4(0.f, 0.f, 0.f, 0.f);
    if (pe < E) {
      const int node = idx[2 * pe + (c0 >> 6)];
      v = *(const float4*)(emb + (size_t)node * D + (c0 & 63));
    }
    sPair[p * TD + c0 + 0] = (_Float16)v.x;
    sPair[p * TD + c0 + 1] = (_Float16)v.y;
    sPair[p * TD + c0 + 2] = (_Float16)v.z;
    sPair[p * TD + c0 + 3] = (_Float16)v.w;
  }
  __syncthreads();

  const int row0 = wave * 16;
  const int nCol = lane & 15;

  // ---- GEMM1: h = relu(pair @ Wm1 + bm1) ----
  v16h a0 = load_a_frag(sPair, TD, row0, lane, 0);
  v16h a1 = load_a_frag(sPair, TD, row0, lane, 32);
  v16h a2 = load_a_frag(sPair, TD, row0, lane, 64);
  v16h a3 = load_a_frag(sPair, TD, row0, lane, 96);
#pragma unroll
  for (int nt = 0; nt < 8; ++nt) {
    const float bias = sB1[nt * 16 + nCol];
    v8f c;
#pragma unroll
    for (int j = 0; j < 8; ++j) c[j] = bias;
    c = wmma_step(a0, load_b_frag(sW1, nt * 4 + 0, lane), c);
    c = wmma_step(a1, load_b_frag(sW1, nt * 4 + 1, lane), c);
    c = wmma_step(a2, load_b_frag(sW1, nt * 4 + 2, lane), c);
    c = wmma_step(a3, load_b_frag(sW1, nt * 4 + 3, lane), c);
#pragma unroll
    for (int j = 0; j < 8; ++j) {
      const float v  = c[j] > 0.f ? c[j] : 0.f;
      const int   lr = ((lane >> 4) << 3) + j;
      sH[(row0 + lr) * TD + nt * 16 + nCol] = (_Float16)v;
    }
  }

  // ---- GEMM2: msg = pair + (h @ Wm2 + bm2), scatter-add into agg ----
  v16h h0 = load_a_frag(sH, TD, row0, lane, 0);
  v16h h1 = load_a_frag(sH, TD, row0, lane, 32);
  v16h h2 = load_a_frag(sH, TD, row0, lane, 64);
  v16h h3 = load_a_frag(sH, TD, row0, lane, 96);
#pragma unroll
  for (int nt = 0; nt < 8; ++nt) {
    const float bias = sB2[nt * 16 + nCol];
    v8f c;
#pragma unroll
    for (int j = 0; j < 8; ++j) c[j] = bias;
    c = wmma_step(h0, load_b_frag(sW2, nt * 4 + 0, lane), c);
    c = wmma_step(h1, load_b_frag(sW2, nt * 4 + 1, lane), c);
    c = wmma_step(h2, load_b_frag(sW2, nt * 4 + 2, lane), c);
    c = wmma_step(h3, load_b_frag(sW2, nt * 4 + 3, lane), c);
#pragma unroll
    for (int j = 0; j < 8; ++j) {
      const int lr = ((lane >> 4) << 3) + j;
      const int p  = row0 + lr;
      const int pe = pairBase + p;
      if (pe < E) {
        const int   col  = nt * 16 + nCol;
        const float msg  = (float)sPair[p * TD + col] + c[j];
        const int   node = idx[2 * pe + (col >> 6)];
        atomicAdd(agg + (size_t)node * D + (col & 63), msg);
      }
    }
  }
}

// ================= Node update MLP + layernorm + residual =================
__global__ __launch_bounds__(128, 1) void update_kernel(
    const float* __restrict__ embIn, const float* __restrict__ agg,
    const _Float16* __restrict__ wu1f, const float* __restrict__ bu1,
    const _Float16* __restrict__ wu2f, const float* __restrict__ bu2,
    const float* __restrict__ gamma, const float* __restrict__ beta,
    float* __restrict__ embOut, int N) {
  __shared__ __align__(16) _Float16 sW1[TD * TD];  // 32 KB fragment-order
  __shared__ __align__(16) _Float16 sW2[TD * D];   // 16 KB fragment-order (16 frags)
  __shared__ float sB1[TD];
  __shared__ float sB2[D];
  __shared__ float sG[D];
  __shared__ float sBt[D];
  __shared__ __align__(16) _Float16 sX[64 * TD];   // 16 KB concat(emb, agg)
  __shared__ __align__(16) _Float16 sH[64 * TD];   // 16 KB
  __shared__ float sU[64 * D];                     // 16 KB fp32 pre-LN output

  const int tid  = threadIdx.x;
  const int lane = tid & 31;
  const int wave = tid >> 5;

  {
    const uint4* g1 = (const uint4*)wu1f;
    const uint4* g2 = (const uint4*)wu2f;
    uint4* s1 = (uint4*)sW1;
    uint4* s2 = (uint4*)sW2;
#pragma unroll
    for (int i = tid; i < TD * TD / 8; i += 128) s1[i] = g1[i];
#pragma unroll
    for (int i = tid; i < TD * D / 8; i += 128) s2[i] = g2[i];
  }
  if (tid < TD) sB1[tid] = bu1[tid];
  if (tid < D) {
    sB2[tid] = bu2[tid];
    sG[tid]  = gamma[tid];
    sBt[tid] = beta[tid];
  }

  const int nb = blockIdx.x * 64;
  for (int i = tid; i < 64 * 32; i += 128) {
    const int p    = i >> 5;
    const int c0   = (i & 31) << 2;
    const int node = nb + p;
    float4 v = make_float4(0.f, 0.f, 0.f, 0.f);
    if (node < N) {
      const float* src = (c0 < D) ? (embIn + (size_t)node * D + c0)
                                  : (agg + (size_t)node * D + (c0 - D));
      v = *(const float4*)src;
    }
    sX[p * TD + c0 + 0] = (_Float16)v.x;
    sX[p * TD + c0 + 1] = (_Float16)v.y;
    sX[p * TD + c0 + 2] = (_Float16)v.z;
    sX[p * TD + c0 + 3] = (_Float16)v.w;
  }
  __syncthreads();

  const int row0 = wave * 16;
  const int nCol = lane & 15;

  // GEMM1: relu(x @ Wu1 + bu1)
  v16h a0 = load_a_frag(sX, TD, row0, lane, 0);
  v16h a1 = load_a_frag(sX, TD, row0, lane, 32);
  v16h a2 = load_a_frag(sX, TD, row0, lane, 64);
  v16h a3 = load_a_frag(sX, TD, row0, lane, 96);
#pragma unroll
  for (int nt = 0; nt < 8; ++nt) {
    const float bias = sB1[nt * 16 + nCol];
    v8f c;
#pragma unroll
    for (int j = 0; j < 8; ++j) c[j] = bias;
    c = wmma_step(a0, load_b_frag(sW1, nt * 4 + 0, lane), c);
    c = wmma_step(a1, load_b_frag(sW1, nt * 4 + 1, lane), c);
    c = wmma_step(a2, load_b_frag(sW1, nt * 4 + 2, lane), c);
    c = wmma_step(a3, load_b_frag(sW1, nt * 4 + 3, lane), c);
#pragma unroll
    for (int j = 0; j < 8; ++j) {
      const float v  = c[j] > 0.f ? c[j] : 0.f;
      const int   lr = ((lane >> 4) << 3) + j;
      sH[(row0 + lr) * TD + nt * 16 + nCol] = (_Float16)v;
    }
  }

  // GEMM2: u = h @ Wu2 + bu2 (64 output cols -> 4 n-tiles)
  v16h h0 = load_a_frag(sH, TD, row0, lane, 0);
  v16h h1 = load_a_frag(sH, TD, row0, lane, 32);
  v16h h2 = load_a_frag(sH, TD, row0, lane, 64);
  v16h h3 = load_a_frag(sH, TD, row0, lane, 96);
#pragma unroll
  for (int nt = 0; nt < 4; ++nt) {
    const float bias = sB2[nt * 16 + nCol];
    v8f c;
#pragma unroll
    for (int j = 0; j < 8; ++j) c[j] = bias;
    c = wmma_step(h0, load_b_frag(sW2, nt * 4 + 0, lane), c);
    c = wmma_step(h1, load_b_frag(sW2, nt * 4 + 1, lane), c);
    c = wmma_step(h2, load_b_frag(sW2, nt * 4 + 2, lane), c);
    c = wmma_step(h3, load_b_frag(sW2, nt * 4 + 3, lane), c);
#pragma unroll
    for (int j = 0; j < 8; ++j) {
      const int lr = ((lane >> 4) << 3) + j;
      sU[(row0 + lr) * D + nt * 16 + nCol] = c[j];
    }
  }
  __syncthreads();

  // layernorm + residual; 2 lanes per row, 32 cols each
  const int lrow = lane >> 1;
  const int half = lane & 1;
  const int p    = row0 + lrow;
  float s = 0.f, s2 = 0.f;
#pragma unroll
  for (int c = half * 32; c < half * 32 + 32; ++c) {
    const float u = sU[p * D + c];
    s += u;
    s2 += u * u;
  }
  s  += __shfl_xor(s, 1, 32);
  s2 += __shfl_xor(s2, 1, 32);
  const float mu  = s * (1.f / 64.f);
  const float var = s2 * (1.f / 64.f) - mu * mu;
  const float inv = rsqrtf(var + 1e-5f);
  const int   node = nb + p;
  if (node < N) {
#pragma unroll
    for (int c = half * 32; c < half * 32 + 32; ++c) {
      const float u = (sU[p * D + c] - mu) * inv * sG[c] + sBt[c];
      embOut[(size_t)node * D + c] = embIn[(size_t)node * D + c] + u;
    }
  }
}

extern "C" void kernel_launch(void* const* d_in, const int* in_sizes, int n_in,
                              void* d_out, int out_size, void* d_ws, size_t ws_size,
                              hipStream_t stream) {
  const float* emb0   = (const float*)d_in[0];
  const int*   relIdx = (const int*)d_in[1];
  const float* Wm1    = (const float*)d_in[2];
  const float* bm1    = (const float*)d_in[3];
  const float* Wm2    = (const float*)d_in[4];
  const float* bm2    = (const float*)d_in[5];
  const float* Wu1    = (const float*)d_in[6];
  const float* bu1    = (const float*)d_in[7];
  const float* Wu2    = (const float*)d_in[8];
  const float* bu2    = (const float*)d_in[9];
  const float* gamma  = (const float*)d_in[10];
  const float* beta   = (const float*)d_in[11];

  const int N = in_sizes[0] / D;          // nodes
  const int R = in_sizes[2] / (TD * TD);  // relations
  const int E = in_sizes[1] / (2 * R);    // pairs per relation

  // workspace layout
  float*    agg  = (float*)d_ws;                   // [N, D]
  float*    emb1 = agg + (size_t)N * D;            // [N, D]
  _Float16* wm1f = (_Float16*)(emb1 + (size_t)N * D);
  _Float16* wm2f = wm1f + (size_t)R * TD * TD;
  _Float16* wu1f = wm2f + (size_t)R * TD * TD;
  _Float16* wu2f = wu1f + (size_t)TD * TD;

  // one-time (per call) weight swizzle into fragment order
  swizzle_weights_kernel<<<dim3(64, R), 256, 0, stream>>>(Wm1, wm1f, TD, 32,
                                                          (long)TD * TD,
                                                          (long)TD * TD);
  swizzle_weights_kernel<<<dim3(64, R), 256, 0, stream>>>(Wm2, wm2f, TD, 32,
                                                          (long)TD * TD,
                                                          (long)TD * TD);
  swizzle_weights_kernel<<<dim3(64, 1), 256, 0, stream>>>(Wu1, wu1f, TD, 32, 0, 0);
  swizzle_weights_kernel<<<dim3(32, 1), 256, 0, stream>>>(Wu2, wu2f, D, 16, 0, 0);

  const float* cur = emb0;
  float* outBufs[2] = {emb1, (float*)d_out};
  for (int layer = 0; layer < 2; ++layer) {
    hipMemsetAsync(agg, 0, (size_t)N * D * sizeof(float), stream);
    dim3 gMsg((E + 63) / 64, R);
    msg_scatter_kernel<<<gMsg, 128, 0, stream>>>(cur, relIdx, wm1f, bm1, wm2f, bm2,
                                                 agg, E);
    update_kernel<<<(N + 63) / 64, 128, 0, stream>>>(cur, agg, wu1f, bu1, wu2f, bu2,
                                                     gamma, beta, outBufs[layer], N);
    cur = outBufs[layer];
  }
}
